// GraphSage_7713761263795
// MI455X (gfx1250) — matrix operand order repaired and checked by reading, not compiled, not run on hardware
//
#include <hip/hip_runtime.h>

typedef __attribute__((ext_vector_type(16))) _Float16 v16h;
typedef __attribute__((ext_vector_type(8)))  float    v8f;

#define EPS 1e-5f

// ---------------------------------------------------------------------------
// utility kernels
// ---------------------------------------------------------------------------
__global__ void zero_kernel(float* __restrict__ p, int n) {
  int i = blockIdx.x * blockDim.x + threadIdx.x;
  if (i < n) p[i] = 0.0f;
}

__global__ void degree_kernel(const int* __restrict__ dst, float* __restrict__ deg, int E) {
  int e = blockIdx.x * blockDim.x + threadIdx.x;
  if (e < E) atomicAdd(&deg[dst[e]], 1.0f);
}

__global__ void invdeg_kernel(float* __restrict__ deg, int n) {
  int i = blockIdx.x * blockDim.x + threadIdx.x;
  if (i < n) {
    float d = deg[i];
    deg[i] = (d > 0.0f) ? (1.0f / d) : 0.0f;
  }
}

// 16 threads per edge; each thread handles 4 contiguous features.
// x gather is coalesced (64 floats across 16 lanes); scatter-adds stay in L2.
__global__ void scatter_kernel(const float* __restrict__ x,
                               const int* __restrict__ src,
                               const int* __restrict__ dst,
                               float* __restrict__ agg, int E) {
  int t = blockIdx.x * blockDim.x + threadIdx.x;
  int e = t >> 4;
  if (e >= E) return;
  int c = (t & 15) << 2;
  int s = src[e];
  int d = dst[e];
  const float4 v = *(const float4*)(x + s * 64 + c);
  float* a = agg + d * 64 + c;
  atomicAdd(a + 0, v.x);
  atomicAdd(a + 1, v.y);
  atomicAdd(a + 2, v.z);
  atomicAdd(a + 3, v.w);
}

// ---------------------------------------------------------------------------
// Pack all 8 weight matrices (w_res, w_l[0..2], w_r[0..2], w_fc) from f32
// row-major [64x64] into the per-lane f16 WMMA B-matrix layout:
//   per (nt,kb) 32x16 block, lane L holds column n = nt*16 + (L&15),
//   f16 element j holds K = kb*32 + (j>>3)*16 + (L>>4)*8 + (j&7)  (B striping per ISA §7.12.2)
// Stored so the GEMM reads one contiguous v16h (32B) per lane.
// ---------------------------------------------------------------------------
__global__ void pack_kernel(const float* __restrict__ wl,
                            const float* __restrict__ wr,
                            const float* __restrict__ wres,
                            const float* __restrict__ wfc,
                            _Float16* __restrict__ pack) {
  int tid = blockIdx.x * blockDim.x + threadIdx.x;   // 8 * 4096
  if (tid >= 8 * 4096) return;
  int mat  = tid >> 12;
  int rem  = tid & 4095;
  int j    = rem & 15;
  int lane = (rem >> 4) & 31;
  int blk  = rem >> 9;            // nt*2 + kb
  int nt   = blk >> 1;
  int kb   = blk & 1;
  int n = nt * 16 + (lane & 15);
  int K = kb * 32 + (j >> 3) * 16 + (lane >> 4) * 8 + (j & 7);
  const float* W;
  if      (mat == 0) W = wres;
  else if (mat <= 3) W = wl + (mat - 1) * 4096;
  else if (mat <= 6) W = wr + (mat - 4) * 4096;
  else               W = wfc;
  pack[tid] = (_Float16)W[K * 64 + n];
}

// ---------------------------------------------------------------------------
// A-matrix loader: 16-bit A 16x32 layout (ISA §7.12.2):
//   lane L: M = L&15; VGPR0-3 hold K = half*8 + 0..7, VGPR4-7 hold K = 16 + half*8 + 0..7
// rowp points at the 64-float feature row; kb selects K block (0 or 1).
// ---------------------------------------------------------------------------
__device__ __forceinline__ v16h load_a(const float* __restrict__ rowp, int kb, int half, float scale) {
  const float* p = rowp + kb * 32 + half * 8;
  float f[16];
#pragma unroll
  for (int q = 0; q < 8; ++q) f[q] = p[q];
#pragma unroll
  for (int q = 0; q < 8; ++q) f[8 + q] = p[16 + q];
  v16h a;
#pragma unroll
  for (int q = 0; q < 16; ++q) a[q] = (_Float16)(f[q] * scale);
  return a;
}

// sum across each 16-lane half of the wave32
__device__ __forceinline__ float halfred(float v) {
  v += __shfl_xor(v, 1, 32);
  v += __shfl_xor(v, 2, 32);
  v += __shfl_xor(v, 4, 32);
  v += __shfl_xor(v, 8, 32);
  return v;
}

// ---------------------------------------------------------------------------
// Fused SAGE layer: h = (agg*invd) @ Wl + b + x @ Wr ; x_out = relu(LN(h)) + res
// One 16-row tile per wave; 16 x v_wmma_f32_16x16x32_f16 per tile.
// LayerNorm done in registers via half-wave shuffles (no LDS, no barriers).
// ---------------------------------------------------------------------------
__global__ void __launch_bounds__(256)
sage_layer_kernel(const float* __restrict__ x,
                  const float* __restrict__ agg,
                  const float* __restrict__ invd,
                  const _Float16* __restrict__ packL,
                  const _Float16* __restrict__ packR,
                  const float* __restrict__ bl,
                  const float* __restrict__ gam,
                  const float* __restrict__ bet,
                  const float* __restrict__ res,
                  float* __restrict__ out, int ntiles) {
  int wave = threadIdx.x >> 5;
  int lane = threadIdx.x & 31;
  int tile = blockIdx.x * 8 + wave;
  if (tile >= ntiles) return;              // wave-uniform: EXEC stays all-ones
  int m0    = tile * 16;
  int half  = lane >> 4;
  int col16 = lane & 15;

  int arow = m0 + col16;                   // A-matrix row for this lane
  float sc = invd[arow];
  const float* xr = x   + arow * 64;
  const float* ar = agg + arow * 64;
  v16h ax0 = load_a(xr, 0, half, 1.0f);
  v16h ax1 = load_a(xr, 1, half, 1.0f);
  v16h ag0 = load_a(ar, 0, half, sc);
  v16h ag1 = load_a(ar, 1, half, sc);

  const v16h* BL = (const v16h*)packL;
  const v16h* BR = (const v16h*)packR;

  v8f acc[4];
#pragma unroll
  for (int nt = 0; nt < 4; ++nt) {
    v8f c = {};
    c = __builtin_amdgcn_wmma_f32_16x16x32_f16(false, ag0, false, BL[(nt*2+0)*32 + lane], (short)0, c, false, false);
    c = __builtin_amdgcn_wmma_f32_16x16x32_f16(false, ag1, false, BL[(nt*2+1)*32 + lane], (short)0, c, false, false);
    c = __builtin_amdgcn_wmma_f32_16x16x32_f16(false, ax0, false, BR[(nt*2+0)*32 + lane], (short)0, c, false, false);
    c = __builtin_amdgcn_wmma_f32_16x16x32_f16(false, ax1, false, BR[(nt*2+1)*32 + lane], (short)0, c, false, false);
    acc[nt] = c;
  }

  float cb[4], cg[4], cbt[4];
#pragma unroll
  for (int nt = 0; nt < 4; ++nt) {
    int n = nt * 16 + col16;
    cb[nt]  = bl[n];
    cg[nt]  = gam[n];
    cbt[nt] = bet[n];
  }

  // D layout: VGPR r of lane L = row (r + 8*half), col (nt*16 + col16)
#pragma unroll
  for (int r = 0; r < 8; ++r) {
    float hv[4], s = 0.0f, s2 = 0.0f;
#pragma unroll
    for (int nt = 0; nt < 4; ++nt) {
      float h = acc[nt][r] + cb[nt];
      hv[nt] = h;
      s  += h;
      s2 += h * h;
    }
    s  = halfred(s);
    s2 = halfred(s2);
    float mu   = s * (1.0f / 64.0f);
    float var  = s2 * (1.0f / 64.0f) - mu * mu;
    float rstd = rsqrtf(var + EPS);
    int orow = m0 + r + 8 * half;
    const float* rr  = res + orow * 64;
    float*       orp = out + orow * 64;
#pragma unroll
    for (int nt = 0; nt < 4; ++nt) {
      float v = (hv[nt] - mu) * rstd * cg[nt] + cbt[nt];
      v = fmaxf(v, 0.0f);
      orp[nt * 16 + col16] = v + rr[nt * 16 + col16];
    }
  }
}

// ---------------------------------------------------------------------------
// out = x @ W + b  (residual projection & final FC)
// ---------------------------------------------------------------------------
__global__ void __launch_bounds__(256)
gemm_bias_kernel(const float* __restrict__ x,
                 const _Float16* __restrict__ packB,
                 const float* __restrict__ bias,
                 float* __restrict__ out, int ntiles) {
  int wave = threadIdx.x >> 5;
  int lane = threadIdx.x & 31;
  int tile = blockIdx.x * 8 + wave;
  if (tile >= ntiles) return;
  int m0    = tile * 16;
  int half  = lane >> 4;
  int col16 = lane & 15;

  const float* xr = x + (m0 + col16) * 64;
  v16h a0 = load_a(xr, 0, half, 1.0f);
  v16h a1 = load_a(xr, 1, half, 1.0f);
  const v16h* B = (const v16h*)packB;

#pragma unroll
  for (int nt = 0; nt < 4; ++nt) {
    v8f c = {};
    c = __builtin_amdgcn_wmma_f32_16x16x32_f16(false, a0, false, B[(nt*2+0)*32 + lane], (short)0, c, false, false);
    c = __builtin_amdgcn_wmma_f32_16x16x32_f16(false, a1, false, B[(nt*2+1)*32 + lane], (short)0, c, false, false);
    float bv = bias[nt * 16 + col16];
#pragma unroll
    for (int r = 0; r < 8; ++r) {
      int orow = m0 + r + 8 * half;
      out[orow * 64 + nt * 16 + col16] = c[r] + bv;
    }
  }
}

// ---------------------------------------------------------------------------
extern "C" void kernel_launch(void* const* d_in, const int* in_sizes, int n_in,
                              void* d_out, int out_size, void* d_ws, size_t ws_size,
                              hipStream_t stream) {
  const float* x     = (const float*)d_in[0];
  const int*   esrc  = (const int*)  d_in[1];
  const int*   edst  = (const int*)  d_in[2];
  const float* w_l   = (const float*)d_in[3];
  const float* b_l   = (const float*)d_in[4];
  const float* w_r   = (const float*)d_in[5];
  const float* gamma = (const float*)d_in[6];
  const float* beta  = (const float*)d_in[7];
  const float* w_res = (const float*)d_in[8];
  const float* b_res = (const float*)d_in[9];
  const float* w_fc  = (const float*)d_in[10];
  const float* b_fc  = (const float*)d_in[11];
  float* out = (float*)d_out;

  const int N = in_sizes[0] / 64;          // 100000
  const int E = in_sizes[1];               // 1600000
  const int ntiles = (N + 15) / 16;        // 6250 (N divisible by 16)
  const int nf = N * 64;

  // workspace layout (floats)
  float* ws   = (float*)d_ws;
  float* invd = ws;                                   // [N]
  float* agg  = ws + 100352;                          // [N*64] (aligned pad)
  float* xa   = agg + nf;                             // [N*64]
  float* xb   = xa + nf;                              // [N*64]; aliases residual buf
  _Float16* pack = (_Float16*)(xb + nf);              // 8 * 4096 halves (32B aligned)

  const int T = 256;
  dim3 blk(T);

  // degree -> inv_deg
  zero_kernel<<<(N + T - 1) / T, blk, 0, stream>>>(invd, N);
  pack_kernel<<<(8 * 4096 + T - 1) / T, blk, 0, stream>>>(w_l, w_r, w_res, w_fc, pack);
  degree_kernel<<<(E + T - 1) / T, blk, 0, stream>>>(edst, invd, E);
  invdeg_kernel<<<(N + T - 1) / T, blk, 0, stream>>>(invd, N);

  const int gemm_blocks = (ntiles + 7) / 8;

  // residual projection (layer 0 only): xb = x @ w_res + b_res
  gemm_bias_kernel<<<gemm_blocks, blk, 0, stream>>>(x, pack + 0 * 4096, b_res, xb, ntiles);

  const int scat_blocks = (E * 16 + T - 1) / T;
  const int zero_blocks = (nf + T - 1) / T;

  // layer 0: x -> xa (residual = xb)
  zero_kernel<<<zero_blocks, blk, 0, stream>>>(agg, nf);
  scatter_kernel<<<scat_blocks, blk, 0, stream>>>(x, esrc, edst, agg, E);
  sage_layer_kernel<<<gemm_blocks, blk, 0, stream>>>(
      x, agg, invd, pack + 1 * 4096, pack + 4 * 4096,
      b_l + 0 * 64, gamma + 0 * 64, beta + 0 * 64, xb, xa, ntiles);

  // layer 1: xa -> xb (residual = xa)
  zero_kernel<<<zero_blocks, blk, 0, stream>>>(agg, nf);
  scatter_kernel<<<scat_blocks, blk, 0, stream>>>(xa, esrc, edst, agg, E);
  sage_layer_kernel<<<gemm_blocks, blk, 0, stream>>>(
      xa, agg, invd, pack + 2 * 4096, pack + 5 * 4096,
      b_l + 1 * 64, gamma + 1 * 64, beta + 1 * 64, xa, xb, ntiles);

  // layer 2: xb -> xa (residual = xb)
  zero_kernel<<<zero_blocks, blk, 0, stream>>>(agg, nf);
  scatter_kernel<<<scat_blocks, blk, 0, stream>>>(xb, esrc, edst, agg, E);
  sage_layer_kernel<<<gemm_blocks, blk, 0, stream>>>(
      xb, agg, invd, pack + 3 * 4096, pack + 6 * 4096,
      b_l + 2 * 64, gamma + 2 * 64, beta + 2 * 64, xb, xa, ntiles);

  // final fc: out = xa @ w_fc + b_fc
  gemm_bias_kernel<<<gemm_blocks, blk, 0, stream>>>(xa, pack + 7 * 4096, b_fc, out, ntiles);
}
